// FDM_3899830304921
// MI455X (gfx1250) — matrix operand order repaired and checked by reading, not compiled, not run on hardware
//
#include <hip/hip_runtime.h>
#include <hip/hip_bf16.h>

// FDM: batched cosine-distance matrix + dual softmax + two re-projection GEMMs.
// All three GEMMs run on v_wmma_f32_16x16x32_bf16 (wave32 CDNA5 WMMA).
// Loaders: branch-free (clamped addresses + value masks), b128 global loads,
// packed b32/b64 LDS stores, register-prefetch software pipeline.

typedef __attribute__((ext_vector_type(16))) __bf16 v16bf;
typedef __attribute__((ext_vector_type(8)))  float  v8f;

#define B_  32
#define C_  512
#define N_  784
#define LDT 40   // LDS row stride in bf16: 80 B rows, 16B-aligned, conflict-free

// pack two floats into a dword of two bf16
__device__ inline unsigned pack_bf16(float lo, float hi) {
  union { __bf16 h[2]; unsigned u; } p;
  p.h[0] = (__bf16)lo;
  p.h[1] = (__bf16)hi;
  return p.u;
}

// ---- WMMA fragment loaders (ISA 7.12.2 16-bit layouts), forced b128 DS -----
// A (16x32 bf16): lane = 16*h + m ; elems 0..7 -> K = 8h.. ; 8..15 -> K = 16+8h..
__device__ inline v16bf frag_a(const __bf16* base, int lane) {
  int m = lane & 15, h = lane >> 4;
  const char* p = (const char*)(base + m * LDT + 8 * h);
  union { uint4 q[2]; v16bf v; } u;
  u.q[0] = *(const uint4*)(p);        // K = 8h .. 8h+7
  u.q[1] = *(const uint4*)(p + 32);   // K = 16+8h .. 16+8h+7
  return u.v;
}
// B (32x16 bf16): lane = 16*h + n ; elems 0..7 -> K = 8h.. ; 8..15 -> K = 16+8h..
__device__ inline v16bf frag_b(const __bf16* base, int lane) {
  int n = lane & 15, h = lane >> 4;
  const char* p = (const char*)(base + n * LDT + 8 * h);
  union { uint4 q[2]; v16bf v; } u;
  u.q[0] = *(const uint4*)(p);
  u.q[1] = *(const uint4*)(p + 32);
  return u.v;
}

// ---- Pass 1: inverse L2 norms over channels --------------------------------
__global__ void norms_kernel(const float* __restrict__ fm, float* __restrict__ inv) {
  int n = blockIdx.x * blockDim.x + threadIdx.x;
  int b = blockIdx.y;
  if (n >= N_) return;
  const float* p = fm + (size_t)b * C_ * N_ + n;
  float acc = 0.f;
  for (int c = 0; c < C_; ++c) { float v = p[(size_t)c * N_]; acc = fmaf(v, v, acc); }
  inv[b * N_ + n] = 1.0f / fmaxf(sqrtf(acc), 1e-12f);
}

// ---- Pass 2: M[b,n1,n2] = -(F1n^T @ F2n), K = C = 512 ----------------------
__global__ __launch_bounds__(256) void gemm1_kernel(
    const float* __restrict__ fm1, const float* __restrict__ fm2,
    const float* __restrict__ inv1, const float* __restrict__ inv2,
    float* __restrict__ Mout) {
  __shared__ __bf16 Asm[128 * LDT];
  __shared__ __bf16 Bsm[128 * LDT];
  int b = blockIdx.z;
  int row0 = blockIdx.x * 128;   // n1
  int col0 = blockIdx.y * 128;   // n2
  int t = threadIdx.x, lane = t & 31, wave = t >> 5;
  const float* f1 = fm1 + (size_t)b * C_ * N_;
  const float* f2 = fm2 + (size_t)b * C_ * N_;

  // loader geometry: 4 spatial rows x (2 adjacent c) x 2 passes per thread
  int nl4 = (t & 31) * 4;        // spatial base inside tile (mult of 4)
  int cp2 = (t >> 5) * 2;        // c-pair base: 0,2,..,14
  int arow = row0 + nl4, brow = col0 + nl4;
  bool ain = arow < N_, bin = brow < N_;   // all-or-nothing (784 % 4 == 0)
  int arowc = ain ? arow : 0, browc = bin ? brow : 0;
  float4 s1v = ain ? *(const float4*)&inv1[b * N_ + arow] : make_float4(0.f, 0.f, 0.f, 0.f);
  float4 s2v = bin ? *(const float4*)&inv2[b * N_ + brow] : make_float4(0.f, 0.f, 0.f, 0.f);
  const float* sv1 = (const float*)&s1v;
  const float* sv2 = (const float*)&s2v;

  int wr = (wave & 3) * 32;      // wave's 32-row strip
  int wc = (wave >> 2) * 64;     // wave's 64-col strip

  v8f acc[2][4] = {};
  float4 ar[4], br[4];

  // prefetch kb = 0
#pragma unroll
  for (int p = 0; p < 2; ++p) {
    int c = cp2 + 16 * p;
    ar[2 * p + 0] = *(const float4*)&f1[(size_t)(c + 0) * N_ + arowc];
    ar[2 * p + 1] = *(const float4*)&f1[(size_t)(c + 1) * N_ + arowc];
    br[2 * p + 0] = *(const float4*)&f2[(size_t)(c + 0) * N_ + browc];
    br[2 * p + 1] = *(const float4*)&f2[(size_t)(c + 1) * N_ + browc];
  }

  for (int kb = 0; kb < C_ / 32; ++kb) {
    // convert + pack + store previous fetch to LDS ([spatial][k], pairs of k)
#pragma unroll
    for (int p = 0; p < 2; ++p) {
      int c = cp2 + 16 * p;
      const float* a0 = (const float*)&ar[2 * p + 0];
      const float* a1 = (const float*)&ar[2 * p + 1];
      const float* b0 = (const float*)&br[2 * p + 0];
      const float* b1 = (const float*)&br[2 * p + 1];
#pragma unroll
      for (int i = 0; i < 4; ++i) {
        *(unsigned*)&Asm[(nl4 + i) * LDT + c] = pack_bf16(a0[i] * sv1[i], a1[i] * sv1[i]);
        *(unsigned*)&Bsm[(nl4 + i) * LDT + c] = pack_bf16(b0[i] * sv2[i], b1[i] * sv2[i]);
      }
    }
    __syncthreads();
    // prefetch next tile (overlaps with WMMA below)
    if (kb + 1 < C_ / 32) {
      int k0 = (kb + 1) * 32;
#pragma unroll
      for (int p = 0; p < 2; ++p) {
        int c = k0 + cp2 + 16 * p;
        ar[2 * p + 0] = *(const float4*)&f1[(size_t)(c + 0) * N_ + arowc];
        ar[2 * p + 1] = *(const float4*)&f1[(size_t)(c + 1) * N_ + arowc];
        br[2 * p + 0] = *(const float4*)&f2[(size_t)(c + 0) * N_ + browc];
        br[2 * p + 1] = *(const float4*)&f2[(size_t)(c + 1) * N_ + browc];
      }
    }
    v16bf afr[2], bfr[4];
#pragma unroll
    for (int i = 0; i < 2; ++i) afr[i] = frag_a(&Asm[(wr + 16 * i) * LDT], lane);
#pragma unroll
    for (int j = 0; j < 4; ++j) bfr[j] = frag_b(&Bsm[(wc + 16 * j) * LDT], lane);
#pragma unroll
    for (int i = 0; i < 2; ++i)
#pragma unroll
      for (int j = 0; j < 4; ++j)
        acc[i][j] = __builtin_amdgcn_wmma_f32_16x16x32_bf16(
            false, afr[i], false, bfr[j], (short)0, acc[i][j], false, false);
    __syncthreads();
  }
  int nloc = lane & 15, h = lane >> 4;
#pragma unroll
  for (int i = 0; i < 2; ++i)
#pragma unroll
    for (int j = 0; j < 4; ++j) {
      int col = col0 + wc + 16 * j + nloc;
      if (col >= N_) continue;
#pragma unroll
      for (int r = 0; r < 8; ++r) {
        int row = row0 + wr + 16 * i + r + 8 * h;
        if (row < N_)
          Mout[((size_t)b * N_ + row) * N_ + col] = -acc[i][j][r];
      }
    }
}

// ---- Pass 3a: per-row max/sumexp (softmax over axis 2) ---------------------
__global__ __launch_bounds__(256) void rowstats_kernel(
    const float* __restrict__ Mm, float* __restrict__ rmax, float* __restrict__ rsum) {
  int wave = threadIdx.x >> 5, lane = threadIdx.x & 31;
  int row = blockIdx.x * 8 + wave;            // in [0, B_*N_)
  if (row >= B_ * N_) return;
  const float* p = Mm + (size_t)row * N_;
  float vals[25];
#pragma unroll
  for (int i = 0; i < 25; ++i) {
    int m = lane + i * 32;
    vals[i] = (m < N_) ? p[m] : -__builtin_inff();
  }
  float mx = -__builtin_inff();
#pragma unroll
  for (int i = 0; i < 25; ++i) mx = fmaxf(mx, vals[i]);
  for (int o = 16; o; o >>= 1) mx = fmaxf(mx, __shfl_xor(mx, o, 32));
  float s = 0.f;
#pragma unroll
  for (int i = 0; i < 25; ++i) s += __expf(vals[i] - mx);
  for (int o = 16; o; o >>= 1) s += __shfl_xor(s, o, 32);
  if (lane == 0) { rmax[row] = mx; rsum[row] = s; }
}

// ---- Pass 3b: per-column max/sumexp (softmax over axis 1), online ----------
__global__ void colstats_kernel(
    const float* __restrict__ Mm, float* __restrict__ cmax, float* __restrict__ csum) {
  int m = blockIdx.x * blockDim.x + threadIdx.x;
  int b = blockIdx.y;
  if (m >= N_) return;
  const float* p = Mm + (size_t)b * N_ * N_ + m;
  float mx = -__builtin_inff(), s = 0.f;
  for (int n = 0; n < N_; ++n) {
    float x  = p[(size_t)n * N_];
    float nm = fmaxf(mx, x);
    s  = s * __expf(mx - nm) + __expf(x - nm);
    mx = nm;
  }
  cmax[b * N_ + m] = mx;
  csum[b * N_ + m] = s;
}

// ---- Pass 4: P1 = softmax(M,axis1) bf16 ; Q = softmax(M,axis2)^T bf16 ------
__global__ __launch_bounds__(256) void softmax_kernel(
    const float* __restrict__ Mm,
    const float* __restrict__ rmax, const float* __restrict__ rsum,
    const float* __restrict__ cmax, const float* __restrict__ csum,
    __bf16* __restrict__ P1, __bf16* __restrict__ Q) {
  __shared__ float T[32][33];
  int b = blockIdx.z;
  int n0 = blockIdx.x * 32, m0 = blockIdx.y * 32;
  int tx = threadIdx.x & 31, ty = threadIdx.x >> 5;
  const float* Mb = Mm + (size_t)b * N_ * N_;
#pragma unroll
  for (int rr = 0; rr < 4; ++rr) {
    int r = ty + rr * 8;
    int n = n0 + r, m = m0 + tx;
    float q = 0.f;
    if (n < N_ && m < N_) {
      float x = Mb[(size_t)n * N_ + m];
      float p = __expf(x - cmax[b * N_ + m]) / csum[b * N_ + m];
      P1[((size_t)b * N_ + n) * N_ + m] = (__bf16)p;
      q = __expf(x - rmax[b * N_ + n]) / rsum[b * N_ + n];
    }
    T[r][tx] = q;
  }
  __syncthreads();
#pragma unroll
  for (int rr = 0; rr < 4; ++rr) {
    int r = ty + rr * 8;
    int qm = m0 + r, qn = n0 + tx;
    if (qm < N_ && qn < N_)
      Q[((size_t)b * N_ + qm) * N_ + qn] = (__bf16)T[tx][r];
  }
}

// ---- Pass 5: out[b,c,m] = scale * sum_k A[b,c,k] * Bw[b,k,m], K = 784 ------
__global__ __launch_bounds__(256) void gemm2_kernel(
    const float* __restrict__ A, const __bf16* __restrict__ Bw,
    float* __restrict__ out, float scale) {
  __shared__ __bf16 Asm[128 * LDT];
  __shared__ __bf16 Bsm[128 * LDT];
  int b = blockIdx.z;
  int c0 = blockIdx.x * 128;    // 4 blocks cover C_=512 exactly
  int m0 = blockIdx.y * 128;
  int t = threadIdx.x, lane = t & 31, wave = t >> 5;
  const float*  Ab = A  + (size_t)b * C_ * N_;
  const __bf16* Bb = Bw + (size_t)b * N_ * N_;

  // A loader: float4 along k; thread -> k4 (0..28 step 4), 4 c-rows (stride 32)
  int k4   = (t & 7) * 4;
  int crow = t >> 3;            // 0..31
  // B loader: uint4 (8 bf16) along m; thread -> 8 m-cols, 2 adjacent k-rows
  int mB   = (t & 15) * 8;      // 0..120
  int kp2  = (t >> 4) * 2;      // 0..30
  bool mIn = (m0 + mB) < N_;    // all-or-nothing (784 % 8 == 0)
  size_t mIdx = mIn ? (size_t)(m0 + mB) : 0;

  int wr = (wave & 3) * 32, wc = (wave >> 2) * 64;

  v8f acc[2][4] = {};
  float4 ar[4];
  uint4  br0, br1;

  // prefetch kb = 0  (k always in range for kb < 24)
#pragma unroll
  for (int p = 0; p < 4; ++p)
    ar[p] = *(const float4*)&Ab[(size_t)(c0 + crow + 32 * p) * N_ + k4];
  br0 = *(const uint4*)&Bb[(size_t)(kp2 + 0) * N_ + mIdx];
  br1 = *(const uint4*)&Bb[(size_t)(kp2 + 1) * N_ + mIdx];

  for (int kb = 0; kb < 25; ++kb) {          // 25*32 = 800 >= 784, zero-padded
    // masks for this tile's k positions
    float    km = (kb * 32 + k4) < N_ ? 1.f : 0.f;        // float4 all-in/out
    unsigned z0 = (kb * 32 + kp2 + 0) < N_ ? ~0u : 0u;
    unsigned z1 = (kb * 32 + kp2 + 1) < N_ ? ~0u : 0u;
    // store A: pack 4 k-values as 2 dwords -> one b64 store (8B-aligned)
#pragma unroll
    for (int p = 0; p < 4; ++p) {
      const float* v = (const float*)&ar[p];
      uint2 pk;
      pk.x = pack_bf16(v[0] * km, v[1] * km);
      pk.y = pack_bf16(v[2] * km, v[3] * km);
      *(uint2*)&Asm[(crow + 32 * p) * LDT + k4] = pk;
    }
    // store B: transpose 2x8 bf16 micro-tile into [m][k] pairs
#pragma unroll
    for (int jj = 0; jj < 4; ++jj) {
      unsigned w0 = ((const unsigned*)&br0)[jj] & z0;   // row k, cols m+2jj, m+2jj+1
      unsigned w1 = ((const unsigned*)&br1)[jj] & z1;   // row k+1
      *(unsigned*)&Bsm[(mB + 2 * jj + 0) * LDT + kp2] = (w0 & 0xffffu) | (w1 << 16);
      *(unsigned*)&Bsm[(mB + 2 * jj + 1) * LDT + kp2] = (w0 >> 16) | (w1 & 0xffff0000u);
    }
    __syncthreads();
    // prefetch next tile (clamped addresses; overlaps with WMMA below)
    if (kb + 1 < 25) {
      int k0 = (kb + 1) * 32;
      int kk = k0 + k4;        size_t kc  = (kk < N_) ? (size_t)kk : 0;
      int kr = k0 + kp2;       size_t kr0 = (kr + 0 < N_) ? (size_t)(kr + 0) : 0;
                               size_t kr1 = (kr + 1 < N_) ? (size_t)(kr + 1) : 0;
#pragma unroll
      for (int p = 0; p < 4; ++p)
        ar[p] = *(const float4*)&Ab[(size_t)(c0 + crow + 32 * p) * N_ + kc];
      br0 = *(const uint4*)&Bb[kr0 * N_ + mIdx];
      br1 = *(const uint4*)&Bb[kr1 * N_ + mIdx];
    }
    v16bf afr[2], bfr[4];
#pragma unroll
    for (int i = 0; i < 2; ++i) afr[i] = frag_a(&Asm[(wr + 16 * i) * LDT], lane);
#pragma unroll
    for (int j = 0; j < 4; ++j) bfr[j] = frag_b(&Bsm[(wc + 16 * j) * LDT], lane);
#pragma unroll
    for (int i = 0; i < 2; ++i)
#pragma unroll
      for (int j = 0; j < 4; ++j)
        acc[i][j] = __builtin_amdgcn_wmma_f32_16x16x32_bf16(
            false, afr[i], false, bfr[j], (short)0, acc[i][j], false, false);
    __syncthreads();
  }
  int nloc = lane & 15, h = lane >> 4;
#pragma unroll
  for (int i = 0; i < 2; ++i)
#pragma unroll
    for (int j = 0; j < 4; ++j) {
      int col = m0 + wc + 16 * j + nloc;
      if (col >= N_) continue;
#pragma unroll
      for (int r = 0; r < 8; ++r) {
        int row = c0 + wr + 16 * i + r + 8 * h;   // always < 512
        out[((size_t)b * C_ + row) * N_ + col] = scale * acc[i][j][r];
      }
    }
}

// ---- Host-side launcher ----------------------------------------------------
extern "C" void kernel_launch(void* const* d_in, const int* in_sizes, int n_in,
                              void* d_out, int out_size, void* d_ws, size_t ws_size,
                              hipStream_t stream) {
  const float* fm1 = (const float*)d_in[0];
  const float* fm2 = (const float*)d_in[1];
  float* out = (float*)d_out;

  char* ws = (char*)d_ws;
  const size_t BN  = (size_t)B_ * N_ * sizeof(float);            // 100,352 B
  const size_t BNN = (size_t)B_ * N_ * N_;
  float*  inv1 = (float*)(ws + 0 * BN);
  float*  inv2 = (float*)(ws + 1 * BN);
  float*  rmax = (float*)(ws + 2 * BN);
  float*  rsum = (float*)(ws + 3 * BN);
  float*  cmax = (float*)(ws + 4 * BN);
  float*  csum = (float*)(ws + 5 * BN);
  float*  Mbuf = (float*)(ws + 6 * BN);                           // 78.7 MB
  __bf16* P1   = (__bf16*)(ws + 6 * BN + BNN * sizeof(float));    // 39.3 MB
  __bf16* Q    = (__bf16*)(ws + 6 * BN + BNN * sizeof(float) + BNN * sizeof(__bf16));

  norms_kernel<<<dim3(7, B_), 128, 0, stream>>>(fm1, inv1);
  norms_kernel<<<dim3(7, B_), 128, 0, stream>>>(fm2, inv2);

  gemm1_kernel<<<dim3(7, 7, B_), 256, 0, stream>>>(fm1, fm2, inv1, inv2, Mbuf);

  rowstats_kernel<<<dim3((B_ * N_) / 8), 256, 0, stream>>>(Mbuf, rmax, rsum);
  colstats_kernel<<<dim3(7, B_), 128, 0, stream>>>(Mbuf, cmax, csum);

  softmax_kernel<<<dim3(25, 25, B_), 256, 0, stream>>>(Mbuf, rmax, rsum, cmax, csum, P1, Q);

  // new_fm1 = 0.001 * F2 @ Q   (first half of d_out)
  gemm2_kernel<<<dim3(4, 7, B_), 256, 0, stream>>>(fm2, Q, out, 0.001f);
  // new_fm2 = 0.001 * F1 @ P1  (second half of d_out)
  gemm2_kernel<<<dim3(4, 7, B_), 256, 0, stream>>>(fm1, P1, out + (size_t)B_ * C_ * N_, 0.001f);
}
